// MultiHeadSelfAttention_79259326480731
// MI455X (gfx1250) — matrix operand order repaired and checked by reading, not compiled, not run on hardware
//
#include <hip/hip_runtime.h>
#include <hip/hip_bf16.h>

// ---------------------------------------------------------------------------
// CDNA5 (gfx1250) wave32 WMMA types
// ---------------------------------------------------------------------------
typedef __attribute__((ext_vector_type(16))) __bf16 v16bf;
typedef __attribute__((ext_vector_type(8)))  __bf16 v8bf;
typedef __attribute__((ext_vector_type(8)))  float  v8f;

#define WMMA_BF16(a, b, c) \
  __builtin_amdgcn_wmma_f32_16x16x32_bf16(false, (a), false, (b), (short)0, (c), false, false)

// softmax temperature (1/sqrt(32)) folded together with log2(e) so the whole
// online softmax runs in the exp2 domain (bare v_exp_f32, no extra muls).
#define QSCALE 0.25508681602824833f

// ---------------------------------------------------------------------------
// Fragment gathers per CDNA5 ISA 7.12.2 (wave32), fully vectorized: each
// fragment is exactly two 16-byte loads per lane (b128), no scalar gathers.
// ---------------------------------------------------------------------------
static __device__ __forceinline__ v16bf load_a_frag(const __bf16* __restrict__ p, int ld) {
  const int lane = threadIdx.x & 31;
  const __bf16* r = p + (lane & 15) * ld + ((lane >> 4) << 3);
  v8bf lo = *(const v8bf*)(r);
  v8bf hi = *(const v8bf*)(r + 16);
  return __builtin_shufflevector(lo, hi, 0, 1, 2, 3, 4, 5, 6, 7,
                                 8, 9, 10, 11, 12, 13, 14, 15);
}

// B 32x16 where element (k,n) lives at p[n*ldn + k] ("transposed" storage).
static __device__ __forceinline__ v16bf load_bT_frag(const __bf16* __restrict__ p, int ldn) {
  const int lane = threadIdx.x & 31;
  const __bf16* c = p + (lane & 15) * ldn + ((lane >> 4) << 3);
  v8bf lo = *(const v8bf*)(c);
  v8bf hi = *(const v8bf*)(c + 16);
  return __builtin_shufflevector(lo, hi, 0, 1, 2, 3, 4, 5, 6, 7,
                                 8, 9, 10, 11, 12, 13, 14, 15);
}

// ---------------------------------------------------------------------------
// Stage 1: projections.  O(256x4096) = W(256x256) * X(256x4096).
// Z is a COMPILE-TIME variant selector -> straight-line epilogues:
//   Z=0 (Q): store scaled by QSCALE (softmax temp + log2e folded)
//   Z=1 (K): plain store
//   Z=2 (V): transposed store VT[c*4096+s], s = m*16+(n>>8), c = n&255
// ---------------------------------------------------------------------------
template <int Z>
__global__ __launch_bounds__(256) void proj_gemm(const float* __restrict__ W,
                                                 const float* __restrict__ X,
                                                 __bf16* __restrict__ O) {
  const int n0 = blockIdx.x * 64;
  const int m0 = blockIdx.y * 64;
  __shared__ __bf16 As[64 * 32];  // As[m*32 + k]
  __shared__ __bf16 Bs[64 * 32];  // transposed: Bs[n*32 + k]
  const int t = threadIdx.x;
  const int wave = t >> 5, lane = t & 31;
  const int wm = (wave >> 1) * 16, wn = (wave & 1) * 32;

  v8f acc0 = {}, acc1 = {};
  for (int k0 = 0; k0 < 256; k0 += 32) {
#pragma unroll
    for (int i = 0; i < 4; ++i) {             // A: 1024 float2 pairs
      const int p = i * 256 + t;
      const int row = p >> 4, col = (p & 15) * 2;
      const float2 w2 = *(const float2*)(W + (m0 + row) * 256 + k0 + col);
      As[row * 32 + col]     = (__bf16)w2.x;
      As[row * 32 + col + 1] = (__bf16)w2.y;
    }
#pragma unroll
    for (int i = 0; i < 8; ++i) {             // B: coalesced read, transposed write
      const int idx = i * 256 + t;
      const int k = idx >> 6, n = idx & 63;
      Bs[n * 32 + k] = (__bf16)X[(k0 + k) * 4096 + n0 + n];
    }
    __syncthreads();
    v16bf a  = load_a_frag(As + wm * 32, 32);
    v16bf b0 = load_bT_frag(Bs + wn * 32, 32);
    v16bf b1 = load_bT_frag(Bs + (wn + 16) * 32, 32);
    acc0 = WMMA_BF16(a, b0, acc0);
    acc1 = WMMA_BF16(a, b1, acc1);
    __syncthreads();
  }
  const int mh = (lane >> 4) << 3;
  const int nn = lane & 15;
#pragma unroll
  for (int v = 0; v < 8; ++v) {
    const int row = m0 + wm + mh + v;
    const int c0 = n0 + wn + nn, c1 = c0 + 16;
    if constexpr (Z == 2) {
      O[(c0 & 255) * 4096 + row * 16 + (c0 >> 8)] = (__bf16)acc0[v];
      O[(c1 & 255) * 4096 + row * 16 + (c1 >> 8)] = (__bf16)acc1[v];
    } else if constexpr (Z == 0) {
      O[row * 4096 + c0] = (__bf16)(acc0[v] * QSCALE);
      O[row * 4096 + c1] = (__bf16)(acc1[v] * QSCALE);
    } else {
      O[row * 4096 + c0] = (__bf16)acc0[v];
      O[row * 4096 + c1] = (__bf16)acc1[v];
    }
  }
}

// ---------------------------------------------------------------------------
// Stage 2: Wcomb = Wv * Wm (256^3 fp32) -- fuses merge-linear + final 1x1 conv.
// ---------------------------------------------------------------------------
__global__ __launch_bounds__(256) void wcomb_kernel(const float* __restrict__ Wv,
                                                    const float* __restrict__ Wm,
                                                    float* __restrict__ Wc) {
  const int o = blockIdx.x, k = threadIdx.x;
  float s = 0.f;
#pragma unroll 4
  for (int c = 0; c < 256; ++c) s = fmaf(Wv[o * 256 + c], Wm[c * 256 + k], s);
  Wc[o * 256 + k] = s;
}

// ---------------------------------------------------------------------------
// Stage 3: flash attention with TRANSPOSED score tiles.
//   T_nt = K_tile(16x32) x Q^T(32x16)  -> query lives in the LANE dim:
//   * row max / row sum are lane-local register trees + one xor-16 combine
//   * T's f32 D-layout elementwise matches the bf16 A-layout needed by P*V:
//       pa0 = bf16(T0) ++ bf16(T1), pa1 = bf16(T2) ++ bf16(T3)
//     -> NO LDS transpose round-trip for P at all.
//   The only cross-orientation fixup: broadcasting per-query corr / 1/l to
//   the accumulator layout via a 64-byte wave-private LDS buffer.
// Block = 4 waves x 16 query rows; no block barriers (wave-private LDS).
// ---------------------------------------------------------------------------
__global__ __launch_bounds__(128) void flash_attn(const __bf16* __restrict__ Q,
                                                  const __bf16* __restrict__ K,
                                                  const __bf16* __restrict__ VT,
                                                  __bf16* __restrict__ Z) {
  const int h    = blockIdx.y;
  const int l0   = blockIdx.x * 64;
  const int wave = threadIdx.x >> 5;
  const int lane = threadIdx.x & 31;
  const int rows = l0 + wave * 16;
  const int mh   = (lane >> 4) << 3;
  const int nn   = lane & 15;

  __shared__ float Cor[4][16];     // per-wave per-query broadcast buffer
  float* cw = Cor[wave];

  // Q^T as the (reused) B operand: B[k=dim, n=query], contiguous per lane.
  const v16bf qT = load_bT_frag(Q + rows * 256 + h * 32, 256);

  v8f az0 = {}, az1 = {};
  float rm = -3.0e38f, rl = 0.f;   // per-lane (per-query) running stats

  for (int s0 = 0; s0 < 4096; s0 += 64) {
    // prefetch next K / V tiles into cache (global_prefetch_b8)
    const int sn = (s0 + 64 < 4096) ? s0 + 64 : 0;
    __builtin_prefetch(K + (sn + lane) * 256 + h * 32, 0, 3);
    __builtin_prefetch(K + (sn + 32 + lane) * 256 + h * 32, 0, 3);
    __builtin_prefetch(VT + (h * 32 + lane) * 4096 + sn, 0, 3);

    // ---- transposed scores: T_nt = K_nt(16x32) x Q^T(32x16) ----
    v8f tc[4];
#pragma unroll
    for (int nt = 0; nt < 4; ++nt) {
      v16bf ka = load_a_frag(K + (s0 + nt * 16) * 256 + h * 32, 256);
      v8f zz = {};
      tc[nt] = WMMA_BF16(ka, qT, zz);
    }
    // ---- lane-local max over this lane's 32 scores, then cross-half ----
    v8f mm;
#pragma unroll
    for (int v = 0; v < 8; ++v)
      mm[v] = fmaxf(fmaxf(tc[0][v], tc[1][v]), fmaxf(tc[2][v], tc[3][v]));
    float tmax = fmaxf(fmaxf(fmaxf(mm[0], mm[1]), fmaxf(mm[2], mm[3])),
                       fmaxf(fmaxf(mm[4], mm[5]), fmaxf(mm[6], mm[7])));
    tmax = fmaxf(tmax, __shfl_xor(tmax, 16));
    const float mnew = fmaxf(rm, tmax);
    const float corr = __builtin_amdgcn_exp2f(rm - mnew);
    rm = mnew;

    // ---- P = exp2(T - mnew), lane-local row sum ----
    float rs = 0.f;
#pragma unroll
    for (int nt = 0; nt < 4; ++nt)
#pragma unroll
      for (int v = 0; v < 8; ++v) {
        const float p = __builtin_amdgcn_exp2f(tc[nt][v] - mnew);
        tc[nt][v] = p;
        rs += p;
      }
    rs += __shfl_xor(rs, 16);
    rl = rl * corr + rs;

    // ---- rescale accumulators: broadcast corr[query] to vgpr layout ----
    cw[nn] = corr;                          // lanes L and L+16 write same value
    const v8f cb = *(const v8f*)(cw + mh);  // corr[mh..mh+7]
#pragma unroll
    for (int v = 0; v < 8; ++v) { az0[v] *= cb[v]; az1[v] *= cb[v]; }

    // ---- P A-fragments assembled straight from registers ----
    v16bf pa0, pa1;
#pragma unroll
    for (int i = 0; i < 8; ++i) {
      pa0[i]     = (__bf16)tc[0][i];
      pa0[8 + i] = (__bf16)tc[1][i];
      pa1[i]     = (__bf16)tc[2][i];
      pa1[8 + i] = (__bf16)tc[3][i];
    }
    // ---- Z += P(16x64) * V(64x32); VT fragments contiguous per lane ----
    v16bf bv;
    bv = load_bT_frag(VT + (h * 32) * 4096 + s0, 4096);             az0 = WMMA_BF16(pa0, bv, az0);
    bv = load_bT_frag(VT + (h * 32) * 4096 + s0 + 32, 4096);        az0 = WMMA_BF16(pa1, bv, az0);
    bv = load_bT_frag(VT + (h * 32 + 16) * 4096 + s0, 4096);        az1 = WMMA_BF16(pa0, bv, az1);
    bv = load_bT_frag(VT + (h * 32 + 16) * 4096 + s0 + 32, 4096);   az1 = WMMA_BF16(pa1, bv, az1);
  }
  // ---- finalize: broadcast 1/l[query] to vgpr layout, store bf16 ----
  cw[nn] = 1.0f / rl;
  const v8f ib = *(const v8f*)(cw + mh);
#pragma unroll
  for (int v = 0; v < 8; ++v) {
    const int row = rows + mh + v;
    Z[row * 256 + h * 32 + nn]      = (__bf16)(az0[v] * ib[v]);
    Z[row * 256 + h * 32 + 16 + nn] = (__bf16)(az1[v] * ib[v]);
  }
}

// ---------------------------------------------------------------------------
// Stage 4: out(256x4096) = Wcomb(256x256) * Z^T + x.
// Z is (4096,256) row-major, so B[k,n] = Z[n*256+k] is contiguous per lane:
// B-fragments load straight from global, only A is staged through LDS.
// ---------------------------------------------------------------------------
__global__ __launch_bounds__(256) void out_gemm(const float* __restrict__ Wc,
                                                const __bf16* __restrict__ Zb,
                                                const float* __restrict__ x,
                                                float* __restrict__ out) {
  const int n0 = blockIdx.x * 64;
  const int m0 = blockIdx.y * 64;
  __shared__ __bf16 As[64 * 32];
  const int t = threadIdx.x;
  const int wave = t >> 5, lane = t & 31;
  const int wm = (wave >> 1) * 16, wn = (wave & 1) * 32;

  v8f acc0 = {}, acc1 = {};
  for (int k0 = 0; k0 < 256; k0 += 32) {
#pragma unroll
    for (int i = 0; i < 4; ++i) {
      const int p = i * 256 + t;
      const int row = p >> 4, col = (p & 15) * 2;
      const float2 w2 = *(const float2*)(Wc + (m0 + row) * 256 + k0 + col);
      As[row * 32 + col]     = (__bf16)w2.x;
      As[row * 32 + col + 1] = (__bf16)w2.y;
    }
    __syncthreads();
    v16bf a  = load_a_frag(As + wm * 32, 32);
    v16bf b0 = load_bT_frag(Zb + (n0 + wn) * 256 + k0, 256);
    v16bf b1 = load_bT_frag(Zb + (n0 + wn + 16) * 256 + k0, 256);
    acc0 = WMMA_BF16(a, b0, acc0);
    acc1 = WMMA_BF16(a, b1, acc1);
    __syncthreads();
  }
  const int mh = (lane >> 4) << 3;
  const int nn = lane & 15;
#pragma unroll
  for (int v = 0; v < 8; ++v) {
    const int row = m0 + wm + mh + v;
    const int col = n0 + wn + nn;
    out[row * 4096 + col]      = acc0[v] + x[row * 4096 + col];
    out[row * 4096 + col + 16] = acc1[v] + x[row * 4096 + col + 16];
  }
}

// ---------------------------------------------------------------------------
// Launch
// ---------------------------------------------------------------------------
extern "C" void kernel_launch(void* const* d_in, const int* in_sizes, int n_in,
                              void* d_out, int out_size, void* d_ws, size_t ws_size,
                              hipStream_t stream) {
  (void)in_sizes; (void)n_in; (void)out_size; (void)ws_size;
  const float* x  = (const float*)d_in[0];
  const float* Wf = (const float*)d_in[1];
  const float* Wg = (const float*)d_in[2];
  const float* Wh = (const float*)d_in[3];
  const float* Wv = (const float*)d_in[4];
  const float* Wm = (const float*)d_in[5];
  float* out = (float*)d_out;

  char* ws = (char*)d_ws;
  __bf16* Qb = (__bf16*)(ws + (0u << 20));   // 2 MB, (4096,256), exp2-domain scaled
  __bf16* Kb = (__bf16*)(ws + (2u << 20));   // 2 MB, (4096,256)
  __bf16* Vt = (__bf16*)(ws + (4u << 20));   // 2 MB, transposed (256,4096)
  __bf16* Zb = (__bf16*)(ws + (6u << 20));   // 2 MB, (4096,256)
  float*  Wc = (float*) (ws + (8u << 20));   // 256 KB

  const dim3 gp(4096 / 64, 256 / 64);
  proj_gemm<0><<<gp, 256, 0, stream>>>(Wf, x, Qb);
  proj_gemm<1><<<gp, 256, 0, stream>>>(Wg, x, Kb);
  proj_gemm<2><<<gp, 256, 0, stream>>>(Wh, x, Vt);
  wcomb_kernel<<<dim3(256), 256, 0, stream>>>(Wv, Wm, Wc);
  flash_attn<<<dim3(4096 / 64, 8), 128, 0, stream>>>(Qb, Kb, Vt, Zb);
  out_gemm<<<gp, 256, 0, stream>>>(Wc, Zb, x, out);
}
